// CausalSelfAttention_55705725829316
// MI455X (gfx1250) — matrix operand order repaired and checked by reading, not compiled, not run on hardware
//
#include <hip/hip_runtime.h>
#include <hip/hip_bf16.h>

// ---------------------------------------------------------------------------
// CausalSelfAttention forward for MI455X (gfx1250), wave32 + WMMA f16.
// Phases: weightT/cvt -> QKV GEMM + RoPE scatter -> flash attention -> proj.
// Matrix math: v_wmma_f32_16x16x32_f16 (f32 accumulate).
// K/V tiles staged to LDS with global_load_async_to_lds_b128 (ASYNCcnt).
// GEMM K-loops are branch-free ping-pong software pipelines (no reg copies).
// ---------------------------------------------------------------------------

typedef __attribute__((ext_vector_type(16))) _Float16 v16h;
typedef __attribute__((ext_vector_type(8)))  _Float16 v8h;
typedef __attribute__((ext_vector_type(8)))  float    v8f;

#define WMMA_F16(A, B, C) \
    __builtin_amdgcn_wmma_f32_16x16x32_f16(false, (A), false, (B), (short)0, (C), false, false)

static constexpr int Bb = 2;
static constexpr int T  = 2048;
static constexpr int C  = 1024;
static constexpr int H  = 16;
static constexpr int D  = 64;
static constexpr int BT = Bb * T;       // 4096
static constexpr int N3 = 3 * C;        // 3072

// ---- CDNA5 async copy: global -> LDS (tracked by ASYNCcnt) -----------------
__device__ __forceinline__ void async_load_b128(const _Float16* g, _Float16* l) {
    // LDS byte address = low 32 bits of the generic pointer (ISA 10.2 aperture).
    uint32_t laddr = (uint32_t)(uintptr_t)l;
    uint64_t gaddr = (uint64_t)(uintptr_t)g;
    asm volatile("global_load_async_to_lds_b128 %0, %1, off"
                 :: "v"(laddr), "v"(gaddr) : "memory");
}
#define S_WAIT_ASYNCCNT(n) asm volatile("s_wait_asynccnt " #n ::: "memory")

// Combine two 8-half chunks into a 16-half A/B fragment register.
__device__ __forceinline__ v16h comb(v8h lo, v8h hi) {
    v16h r;
#pragma unroll
    for (int i = 0; i < 8; ++i) { r[i] = lo[i]; r[8 + i] = hi[i]; }
    return r;
}

// A-fragment (16x32 f16, M=lane%16): lane half h=lane/16 holds K=[8h,8h+8) in
// elements 0..7 and K=[16+8h,16+8h+8) in elements 8..15 (ISA 7.12.2).
__device__ __forceinline__ v16h load_afrag_f16(const _Float16* row, int h16) {
    const _Float16* p = row + h16 * 8;
    return comb(*(const v8h*)(p), *(const v8h*)(p + 16));
}

// Same A-fragment but converting from an f32 source row on the fly.
__device__ __forceinline__ v16h load_afrag_f32(const float* row, int h16) {
    const float* p = row + h16 * 8;
    v16h r;
#pragma unroll
    for (int i = 0; i < 8; ++i) { r[i] = (_Float16)p[i]; r[8 + i] = (_Float16)p[16 + i]; }
    return r;
}

// B-fragment (32x16 f16, N=lane%16): lane half h holds K=[8h,8h+8) in
// elements 0..7 and K=[16+8h,16+8h+8) in elements 8..15.
__device__ __forceinline__ v16h load_bfrag_f16(const _Float16* col, int h16) {
    const _Float16* p = col + h16 * 8;
    return comb(*(const v8h*)(p), *(const v8h*)(p + 16));
}

__device__ __forceinline__ float rowmax16(float v) {
#pragma unroll
    for (int m = 1; m < 16; m <<= 1) v = fmaxf(v, __shfl_xor(v, m));
    return v;
}
__device__ __forceinline__ float rowsum16(float v) {
#pragma unroll
    for (int m = 1; m < 16; m <<= 1) v += __shfl_xor(v, m);
    return v;
}

// ---------------------------------------------------------------------------
// Kernel 0: f32 [K][N] -> f16 [N][K] (weights), LDS tile transpose.
// ---------------------------------------------------------------------------
__global__ __launch_bounds__(256) void transpose_cvt(const float* __restrict__ in,
                                                     _Float16* __restrict__ out,
                                                     int K, int N) {
    __shared__ float tile[32][33];
    const int tx = threadIdx.x, ty = threadIdx.y;     // (32, 8)
    const int n0 = blockIdx.x * 32, k0 = blockIdx.y * 32;
#pragma unroll
    for (int i = 0; i < 4; ++i)
        tile[ty + i * 8][tx] = in[(size_t)(k0 + ty + i * 8) * N + (n0 + tx)];
    __syncthreads();
#pragma unroll
    for (int i = 0; i < 4; ++i)
        out[(size_t)(n0 + ty + i * 8) * K + (k0 + tx)] = (_Float16)tile[tx][ty + i * 8];
}

// ---------------------------------------------------------------------------
// Kernel 1: qkv = x @ w_qkv (M=4096, N=3072, K=1024), WMMA f16.
// WG=256 (8 waves): tile 128(M)x128(N); wave tile 32x64; BK=32; ping-pong
// pipelined fragment fetch (8 WMMAs per 16 b128 loads per K-step).
// Epilogue: RoPE on q,k; scatter q,k -> [B,H,T,D], v -> [B,H,D,T] (f16).
// ---------------------------------------------------------------------------
__global__ __launch_bounds__(256) void qkv_rope(const float* __restrict__ x,
                                                const _Float16* __restrict__ wtq, // [3072][1024] f16
                                                _Float16* __restrict__ qr,
                                                _Float16* __restrict__ kr,
                                                _Float16* __restrict__ vt) {
    const int wid = threadIdx.x >> 5;
    const int lane = threadIdx.x & 31;
    const int lane16 = lane & 15, h16 = lane >> 4;
    const int m0 = blockIdx.x * 128 + (wid & 3) * 32;
    const int n0 = blockIdx.y * 128 + (wid >> 2) * 64;

    v8f acc[2][4];
#pragma unroll
    for (int rr = 0; rr < 2; ++rr)
#pragma unroll
        for (int cc = 0; cc < 4; ++cc) acc[rr][cc] = v8f{};

    const float* arow[2];
#pragma unroll
    for (int rr = 0; rr < 2; ++rr)
        arow[rr] = x + (size_t)(m0 + rr * 16 + lane16) * C;
    const _Float16* wrow[4];
#pragma unroll
    for (int cc = 0; cc < 4; ++cc)
        wrow[cc] = wtq + (size_t)(n0 + cc * 16 + lane16) * C;

    v16h aA[2], aB[2], bA[4], bB[4];
#pragma unroll
    for (int rr = 0; rr < 2; ++rr) aA[rr] = load_afrag_f32(arow[rr], h16);
#pragma unroll
    for (int cc = 0; cc < 4; ++cc) bA[cc] = load_bfrag_f16(wrow[cc], h16);

    for (int k = 0; k < C - 64; k += 64) {
#pragma unroll
        for (int rr = 0; rr < 2; ++rr) aB[rr] = load_afrag_f32(arow[rr] + k + 32, h16);
#pragma unroll
        for (int cc = 0; cc < 4; ++cc) bB[cc] = load_bfrag_f16(wrow[cc] + k + 32, h16);
#pragma unroll
        for (int rr = 0; rr < 2; ++rr)
#pragma unroll
            for (int cc = 0; cc < 4; ++cc) acc[rr][cc] = WMMA_F16(aA[rr], bA[cc], acc[rr][cc]);
#pragma unroll
        for (int rr = 0; rr < 2; ++rr) aA[rr] = load_afrag_f32(arow[rr] + k + 64, h16);
#pragma unroll
        for (int cc = 0; cc < 4; ++cc) bA[cc] = load_bfrag_f16(wrow[cc] + k + 64, h16);
#pragma unroll
        for (int rr = 0; rr < 2; ++rr)
#pragma unroll
            for (int cc = 0; cc < 4; ++cc) acc[rr][cc] = WMMA_F16(aB[rr], bB[cc], acc[rr][cc]);
    }
    // Tail: k = C-64 held in A-set; fetch C-32 then two compute stages.
#pragma unroll
    for (int rr = 0; rr < 2; ++rr) aB[rr] = load_afrag_f32(arow[rr] + C - 32, h16);
#pragma unroll
    for (int cc = 0; cc < 4; ++cc) bB[cc] = load_bfrag_f16(wrow[cc] + C - 32, h16);
#pragma unroll
    for (int rr = 0; rr < 2; ++rr)
#pragma unroll
        for (int cc = 0; cc < 4; ++cc) acc[rr][cc] = WMMA_F16(aA[rr], bA[cc], acc[rr][cc]);
#pragma unroll
    for (int rr = 0; rr < 2; ++rr)
#pragma unroll
        for (int cc = 0; cc < 4; ++cc) acc[rr][cc] = WMMA_F16(aB[rr], bB[cc], acc[rr][cc]);

    // Epilogue: C layout -> element (M = rr*16 + r + 8*h16, N = lane16),
    // RoPE on q/k, scatter to head-major buffers.
#pragma unroll
    for (int cc = 0; cc < 4; ++cc) {
        const int gn = n0 + cc * 16 + lane16;
        const int sec = gn >> 10;          // 0=q 1=k 2=v
        const int ci = gn & (C - 1);
        const int head = ci >> 6;
        const int d = ci & (D - 1);
#pragma unroll
        for (int rr = 0; rr < 2; ++rr)
#pragma unroll
            for (int r = 0; r < 8; ++r) {
                const int gm = m0 + rr * 16 + r + 8 * h16;
                const int bb = gm >> 11, t = gm & (T - 1);
                float v = acc[rr][cc][r];
                float partner = __shfl_xor(v, 1);   // adjacent d = adjacent lane
                float outv;
                if (sec < 2) {
                    const int j = d >> 1;
                    // inv_freq = 10000^(-2j/64) = exp(-ln(1e4) * 2j/64)
                    float invf = __expf(-9.210340372f * (float)(2 * j) * (1.0f / 64.0f));
                    float ang = (float)t * invf;
                    float cs = __cosf(ang), sn = __sinf(ang);
                    bool even = (d & 1) == 0;
                    float e = even ? v : partner;
                    float o = even ? partner : v;
                    outv = even ? (e * cs - o * sn) : (e * sn + o * cs);
                } else {
                    outv = v;
                }
                _Float16 hv = (_Float16)outv;
                const size_t bh = (size_t)bb * H + head;
                if (sec == 0)      qr[(bh * T + t) * D + d] = hv;
                else if (sec == 1) kr[(bh * T + t) * D + d] = hv;
                else               vt[(bh * D + d) * T + t] = hv;   // V^T [D][T]
            }
    }
}

// ---------------------------------------------------------------------------
// Kernel 2: flash attention. WG = 128 thr (4 waves), 16 queries/wave,
// 64 queries/WG, key tiles of 32, K/V tiles async-staged to double-buffered
// LDS (4 async b128 per thread per tile). grid = (T/64, B*H).
// ---------------------------------------------------------------------------
__device__ __forceinline__ void stage_kv_async(const _Float16* kp, const _Float16* vp,
                                               _Float16* kld, _Float16* vld,
                                               int kb, int tid) {
#pragma unroll
    for (int rep = 0; rep < 2; ++rep) {
        const int idx = tid + rep * 128;          // 0..255
        const int krow = idx >> 3, kch = (idx & 7) * 8;
        async_load_b128(kp + (size_t)(kb + krow) * D + kch, kld + krow * D + kch);
        const int vd = idx >> 2, vch = (idx & 3) * 8;
        async_load_b128(vp + (size_t)vd * T + kb + vch, vld + vd * 32 + vch);
    }
}

__global__ __launch_bounds__(128) void attn(const _Float16* __restrict__ qr,
                                            const _Float16* __restrict__ kr,
                                            const _Float16* __restrict__ vt,
                                            _Float16* __restrict__ ybf) {
    __shared__ _Float16 kld[2][32 * D];     // K tile  [key][d]
    __shared__ _Float16 vld[2][D * 32];     // V tile  [d][key]
    __shared__ _Float16 pbuf[4][16 * 32];   // per-wave P staging (C- -> A-layout)

    const int tid = threadIdx.x;
    const int wid = tid >> 5;
    const int lane = tid & 31;
    const int lane16 = lane & 15, h16 = lane >> 4;
    const int qtile = blockIdx.x;
    const int bh = blockIdx.y;
    const int b = bh >> 4, h = bh & (H - 1);

    const _Float16* qp = qr + (size_t)bh * T * D;
    const _Float16* kp = kr + (size_t)bh * T * D;
    const _Float16* vp = vt + (size_t)bh * D * T;

    const int qb = qtile * 64 + wid * 16;

    // Q fragments (16 queries x 64 D), two K=32 chunks, straight from global.
    v16h aq[2];
#pragma unroll
    for (int dk = 0; dk < 2; ++dk)
        aq[dk] = load_afrag_f16(qp + (size_t)(qb + lane16) * D + dk * 32, h16);

    float mrow[8], lrow[8];
    v8f acc[4] = {v8f{}, v8f{}, v8f{}, v8f{}};
#pragma unroll
    for (int r = 0; r < 8; ++r) { mrow[r] = -1e30f; lrow[r] = 0.0f; }

    const float scale = 0.125f;             // 1/sqrt(64)
    const int nkt = 2 * qtile + 2;          // uniform across the WG (causal bound)

    // Prologue: stage tile 0 into buffer 0.
    stage_kv_async(kp, vp, kld[0], vld[0], 0, tid);

    for (int kt = 0; kt < nkt; ++kt) {
        const int kb = kt * 32;
        const int cur = kt & 1, nxt = cur ^ 1;

        // Issue next tile (clamped, redundant on last iter keeps flow uniform).
        int kbn = kb + 32;
        if (kbn > T - 32) kbn = T - 32;
        stage_kv_async(kp, vp, kld[nxt], vld[nxt], kbn, tid);

        // Current tile's 4 async b128 complete once <=4 remain outstanding.
        S_WAIT_ASYNCCNT(0x4);
        __syncthreads();

        const _Float16* kc = kld[cur];
        const _Float16* vc = vld[cur];

        // S = scale * Q K^T for two 16-key subtiles; causal mask per element.
        float sv[2][8];
#pragma unroll
        for (int st = 0; st < 2; ++st) {
            const int key = kb + st * 16 + lane16;   // this lane's B column
            v8f s = {};
#pragma unroll
            for (int dk = 0; dk < 2; ++dk) {
                v16h bk = load_bfrag_f16(kc + (st * 16 + lane16) * D + dk * 32, h16);
                s = WMMA_F16(aq[dk], bk, s);
            }
#pragma unroll
            for (int r = 0; r < 8; ++r) {
                const int q = qb + r + 8 * h16;
                sv[st][r] = (key <= q) ? s[r] * scale : -1e30f;
            }
        }

        // Online softmax update (row stats via 16-lane shuffles).
        float p[2][8], sfac[8];
#pragma unroll
        for (int r = 0; r < 8; ++r) {
            float rm = rowmax16(fmaxf(sv[0][r], sv[1][r]));
            float mn = fmaxf(mrow[r], rm);
            sfac[r] = __expf(mrow[r] - mn);
            float psum = 0.0f;
#pragma unroll
            for (int st = 0; st < 2; ++st) {
                float pv = (sv[st][r] > -5e29f) ? __expf(sv[st][r] - mn) : 0.0f;
                p[st][r] = pv;
                psum += pv;
            }
            lrow[r] = lrow[r] * sfac[r] + rowsum16(psum);
            mrow[r] = mn;
        }
#pragma unroll
        for (int cc = 0; cc < 4; ++cc)
#pragma unroll
            for (int r = 0; r < 8; ++r) acc[cc][r] *= sfac[r];

        // P: C-layout f32 -> A-layout f16 via per-wave LDS transpose.
        _Float16* pb = pbuf[wid];
#pragma unroll
        for (int st = 0; st < 2; ++st)
#pragma unroll
            for (int r = 0; r < 8; ++r)
                pb[(r + 8 * h16) * 32 + st * 16 + lane16] = (_Float16)p[st][r];
        __syncthreads();   // orders pbuf store->load; kld reads already retired

        v16h pa = load_afrag_f16(pb + lane16 * 32, h16);

        // O += P @ V from the LDS V tile.
#pragma unroll
        for (int cc = 0; cc < 4; ++cc) {
            v16h vb = load_bfrag_f16(vc + (cc * 16 + lane16) * 32, h16);
            acc[cc] = WMMA_F16(pa, vb, acc[cc]);
        }
        __syncthreads();   // all waves done with buf[cur] before kt+2 overwrites
    }

    // Normalize and store attention output as f16 [B*T, C] (col = h*64 + d).
    float invl[8];
#pragma unroll
    for (int r = 0; r < 8; ++r) invl[r] = 1.0f / lrow[r];
#pragma unroll
    for (int cc = 0; cc < 4; ++cc) {
        const int col = h * D + cc * 16 + lane16;
#pragma unroll
        for (int r = 0; r < 8; ++r) {
            const int q = qb + r + 8 * h16;
            ybf[((size_t)b * T + q) * C + col] = (_Float16)(acc[cc][r] * invl[r]);
        }
    }
}

// ---------------------------------------------------------------------------
// Kernel 3: out = y @ w_proj + b_proj (M=4096, N=1024, K=1024), f32 out,
// same ping-pong pipelined 32x64 wave tile as kernel 1.
// ---------------------------------------------------------------------------
__global__ __launch_bounds__(256) void proj(const _Float16* __restrict__ ybf,
                                            const _Float16* __restrict__ wtp, // [1024][1024] f16
                                            const float* __restrict__ bias,
                                            float* __restrict__ out) {
    const int wid = threadIdx.x >> 5;
    const int lane = threadIdx.x & 31;
    const int lane16 = lane & 15, h16 = lane >> 4;
    const int m0 = blockIdx.x * 128 + (wid & 3) * 32;
    const int n0 = blockIdx.y * 128 + (wid >> 2) * 64;

    v8f acc[2][4];
#pragma unroll
    for (int rr = 0; rr < 2; ++rr)
#pragma unroll
        for (int cc = 0; cc < 4; ++cc) acc[rr][cc] = v8f{};

    const _Float16* arow[2];
#pragma unroll
    for (int rr = 0; rr < 2; ++rr)
        arow[rr] = ybf + (size_t)(m0 + rr * 16 + lane16) * C;
    const _Float16* wrow[4];
#pragma unroll
    for (int cc = 0; cc < 4; ++cc)
        wrow[cc] = wtp + (size_t)(n0 + cc * 16 + lane16) * C;

    v16h aA[2], aB[2], bA[4], bB[4];
#pragma unroll
    for (int rr = 0; rr < 2; ++rr) aA[rr] = load_afrag_f16(arow[rr], h16);
#pragma unroll
    for (int cc = 0; cc < 4; ++cc) bA[cc] = load_bfrag_f16(wrow[cc], h16);

    for (int k = 0; k < C - 64; k += 64) {
#pragma unroll
        for (int rr = 0; rr < 2; ++rr) aB[rr] = load_afrag_f16(arow[rr] + k + 32, h16);
#pragma unroll
        for (int cc = 0; cc < 4; ++cc) bB[cc] = load_bfrag_f16(wrow[cc] + k + 32, h16);
#pragma unroll
        for (int rr = 0; rr < 2; ++rr)
#pragma unroll
            for (int cc = 0; cc < 4; ++cc) acc[rr][cc] = WMMA_F16(aA[rr], bA[cc], acc[rr][cc]);
#pragma unroll
        for (int rr = 0; rr < 2; ++rr) aA[rr] = load_afrag_f16(arow[rr] + k + 64, h16);
#pragma unroll
        for (int cc = 0; cc < 4; ++cc) bA[cc] = load_bfrag_f16(wrow[cc] + k + 64, h16);
#pragma unroll
        for (int rr = 0; rr < 2; ++rr)
#pragma unroll
            for (int cc = 0; cc < 4; ++cc) acc[rr][cc] = WMMA_F16(aB[rr], bB[cc], acc[rr][cc]);
    }
#pragma unroll
    for (int rr = 0; rr < 2; ++rr) aB[rr] = load_afrag_f16(arow[rr] + C - 32, h16);
#pragma unroll
    for (int cc = 0; cc < 4; ++cc) bB[cc] = load_bfrag_f16(wrow[cc] + C - 32, h16);
#pragma unroll
    for (int rr = 0; rr < 2; ++rr)
#pragma unroll
        for (int cc = 0; cc < 4; ++cc) acc[rr][cc] = WMMA_F16(aA[rr], bA[cc], acc[rr][cc]);
#pragma unroll
    for (int rr = 0; rr < 2; ++rr)
#pragma unroll
        for (int cc = 0; cc < 4; ++cc) acc[rr][cc] = WMMA_F16(aB[rr], bB[cc], acc[rr][cc]);

#pragma unroll
    for (int cc = 0; cc < 4; ++cc) {
        const int gn = n0 + cc * 16 + lane16;
        const float bv = bias[gn];
#pragma unroll
        for (int rr = 0; rr < 2; ++rr)
#pragma unroll
            for (int r = 0; r < 8; ++r) {
                const int gm = m0 + rr * 16 + r + 8 * h16;
                out[(size_t)gm * C + gn] = acc[rr][cc][r] + bv;
            }
    }
}

// ---------------------------------------------------------------------------
extern "C" void kernel_launch(void* const* d_in, const int* in_sizes, int n_in,
                              void* d_out, int out_size, void* d_ws, size_t ws_size,
                              hipStream_t stream) {
    const float* x      = (const float*)d_in[0];
    const float* w_qkv  = (const float*)d_in[1];
    const float* w_proj = (const float*)d_in[2];
    const float* b_proj = (const float*)d_in[3];
    float* out = (float*)d_out;

    // Workspace carve-up (f16 buffers), ~40 MB total.
    _Float16* qr  = (_Float16*)d_ws;                          // [B,H,T,D]
    _Float16* kr  = qr  + (size_t)Bb * H * T * D;             // [B,H,T,D]
    _Float16* vt  = kr  + (size_t)Bb * H * T * D;             // [B,H,D,T]
    _Float16* ybf = vt  + (size_t)Bb * H * T * D;             // [B*T, C]
    _Float16* wtq = ybf + (size_t)BT * C;                     // [3C, C]
    _Float16* wtp = wtq + (size_t)N3 * C;                     // [C, C]

    transpose_cvt<<<dim3(N3 / 32, C / 32), dim3(32, 8), 0, stream>>>(w_qkv, wtq, C, N3);
    transpose_cvt<<<dim3(C / 32, C / 32), dim3(32, 8), 0, stream>>>(w_proj, wtp, C, C);

    qkv_rope<<<dim3(BT / 128, N3 / 128), 256, 0, stream>>>(x, wtq, qr, kr, vt);

    attn<<<dim3(T / 64, Bb * H), 128, 0, stream>>>(qr, kr, vt, ybf);

    proj<<<dim3(BT / 128, C / 128), 256, 0, stream>>>(ybf, wtp, b_proj, out);
}
